// CrossAttentionClassifier_18193481466270
// MI455X (gfx1250) — matrix operand order repaired and checked by reading, not compiled, not run on hardware
//
#include <hip/hip_runtime.h>

// ---------------------------------------------------------------------------
// CDNA5 / gfx1250 fused CrossAttentionClassifier
//   - f16 WMMA (v_wmma_f32_16x16x32_f16) everywhere, fp32 accumulate
//   - len-1 MHA == V-projection; W_v@W_o and biases pre-fused (fp32 prep)
//   - big embed GEMM double-buffered via GLOBAL_LOAD_ASYNC_TO_LDS_B128
// ---------------------------------------------------------------------------

typedef _Float16 h8   __attribute__((ext_vector_type(8)));
typedef _Float16 v16h __attribute__((ext_vector_type(16)));
typedef float    v8f  __attribute__((ext_vector_type(8)));
typedef int      vint4 __attribute__((vector_size(16)));

#define IN_DIM 2056
#define KPAD   2080              // IN_DIM padded to multiple of 32
#define KC     160               // K-chunk (5 ksteps of 32)
#define NCHUNK 13                // 2080 / 160
#define ROWS   64                // rows per workgroup
#define XAF_LD 164               // KC + 4 f32 (656B row stride, bank-safe)
#define H_LD   264               // 256 + 8 halfs
#define CB_LD  520               // 512 + 8 halfs
#define H2_LD  72                // 64 + 8 halfs
#define S32_LD 257               // fp32 staging stride (perfect bank spread)

// ---- LDS byte offsets (time-multiplexed regions) --------------------------
#define OFF_XAF0 0               // 64*164*4 = 41984   (embed A, f32, buf 0)
#define OFF_XAF1 41984           // 41984              (embed A, f32, buf 1)
#define OFF_XB0  83968           // 81920              (embed B, f16, buf 0)
#define OFF_XB1  165888          // 81920 -> 247808    (embed B, f16, buf 1)
#define OFF_S32  0               // 64*257*4 = 65792   (LN staging; inter-phase)
#define OFF_CB   0               // 64*520*2 = 66560   (combined, after embeds)
#define OFF_H1   83968           // 64*264*2 = 33792   (after G5)
#define OFF_H2   117760          // 64*72*2  = 9216
#define OFF_HM   247808          // 33792              (persistent h_mut)
#define OFF_HC   281600          // 33792              (persistent h_ctx)
#define OFF_RS   315392          // 64*2*4   = 512
#define LDS_SZ   315904          // < 320KB

// ---- CDNA5 async global->LDS (guarded; falls back to sync copy) ----------
#if defined(__has_builtin)
#  if __has_builtin(__builtin_amdgcn_global_load_async_to_lds_b128)
#    define HAS_ASYNC 1
#  endif
#endif

__device__ __forceinline__ void async_b128(const void* g, void* l) {
#ifdef HAS_ASYNC
  __builtin_amdgcn_global_load_async_to_lds_b128(
      (__attribute__((address_space(1))) vint4*)g,
      (__attribute__((address_space(3))) vint4*)l, 0, 0);
#else
  *(uint4*)l = *(const uint4*)g;
#endif
}

__device__ __forceinline__ void wait_async0() {
#ifdef HAS_ASYNC
#  if __has_builtin(__builtin_amdgcn_s_wait_asynccnt)
  __builtin_amdgcn_s_wait_asynccnt(0);
#  else
  asm volatile("s_wait_asynccnt 0" ::: "memory");
#  endif
#endif
}

// ---------------------------------------------------------------------------
// Weight repack: fp32 (K x N, row-major) -> f16 WMMA-B tiles.
// Block (kt,nt) = 32x16 sub-matrix; lane = ((k%32)/16)*16 + (n%16),
// each lane owns 16 contiguous halfs (slot = k%16). kt-major block order.
// ---------------------------------------------------------------------------
__global__ void pack_w16(const float* __restrict__ W, _Float16* __restrict__ dst,
                         int K, int KP, int N) {
  int idx = blockIdx.x * blockDim.x + threadIdx.x;
  if (idx >= KP * N) return;
  int k = idx / N, n = idx % N;
  float v = (k < K) ? W[(size_t)k * N + n] : 0.0f;
  int NT = N >> 4;
  int kt = k >> 5, nt = n >> 4;
  int ln = (((k & 31) >> 4) << 4) | (n & 15);
  dst[((size_t)(kt * NT + nt)) * 512 + ln * 16 + (k & 15)] = (_Float16)v;
}

// W_vo[k][n] = sum_j in_w[k][512+j] * out_w[j][n]   (fp32, packed to f16 tiles)
__global__ void fuse_vo_pack(const float* __restrict__ inw,
                             const float* __restrict__ outw,
                             _Float16* __restrict__ dst) {
  int k = blockIdx.x, n = threadIdx.x;        // 256 x 256
  float acc = 0.0f;
  for (int j = 0; j < 256; ++j)
    acc += inw[(size_t)k * 768 + 512 + j] * outw[(size_t)j * 256 + n];
  int kt = k >> 5, nt = n >> 4;
  int ln = (((k & 31) >> 4) << 4) | (n & 15);
  dst[((size_t)(kt * 16 + nt)) * 512 + ln * 16 + (k & 15)] = (_Float16)acc;
}

// b_vo[n] = out_b[n] + sum_j in_b[512+j] * out_w[j][n]
__global__ void fuse_vo_bias(const float* __restrict__ inb,
                             const float* __restrict__ outw,
                             const float* __restrict__ outb,
                             float* __restrict__ dst) {
  int n = threadIdx.x;
  float acc = outb[n];
  for (int j = 0; j < 256; ++j) acc += inb[512 + j] * outw[(size_t)j * 256 + n];
  dst[n] = acc;
}

// ---------------------------------------------------------------------------
// A fragment (16x32) from f16 LDS row-major
__device__ __forceinline__ v16h load_a(const _Float16* base, int stride,
                                       int m, int kb, int kh8) {
  const _Float16* p = base + (size_t)m * stride + kb + kh8;
  h8 lo = *(const h8*)p;
  h8 hi = *(const h8*)(p + 8);
  return __builtin_shufflevector(lo, hi, 0,1,2,3,4,5,6,7,8,9,10,11,12,13,14,15);
}

// A fragment (16x32) from f32 LDS row-major, converted to f16 in regs
__device__ __forceinline__ v16h load_a_f32(const float* base, int strideF,
                                           int m, int kb, int kh8) {
  const float* p = base + (size_t)m * strideF + kb + kh8;
  float4 a0 = *(const float4*)(p);
  float4 a1 = *(const float4*)(p + 4);
  float4 b0 = *(const float4*)(p + 8);
  float4 b1 = *(const float4*)(p + 12);
  v16h v;
  v[0]  = (_Float16)a0.x; v[1]  = (_Float16)a0.y;
  v[2]  = (_Float16)a0.z; v[3]  = (_Float16)a0.w;
  v[4]  = (_Float16)a1.x; v[5]  = (_Float16)a1.y;
  v[6]  = (_Float16)a1.z; v[7]  = (_Float16)a1.w;
  v[8]  = (_Float16)b0.x; v[9]  = (_Float16)b0.y;
  v[10] = (_Float16)b0.z; v[11] = (_Float16)b0.w;
  v[12] = (_Float16)b1.x; v[13] = (_Float16)b1.y;
  v[14] = (_Float16)b1.z; v[15] = (_Float16)b1.w;
  return v;
}

#define WMMA(A, Bv, C) \
  __builtin_amdgcn_wmma_f32_16x16x32_f16(false, (A), false, (Bv), (short)0, (C), false, false)

__global__ __launch_bounds__(256) void fused_net(
    const float* __restrict__ x,
    const _Float16* __restrict__ meP, const float* __restrict__ me_b,
    const float* __restrict__ me_g, const float* __restrict__ me_be,
    const _Float16* __restrict__ ceP, const float* __restrict__ ce_b,
    const float* __restrict__ ce_g, const float* __restrict__ ce_be,
    const _Float16* __restrict__ caP, const float* __restrict__ cab,
    const _Float16* __restrict__ saP, const float* __restrict__ sab,
    const _Float16* __restrict__ c1P, const float* __restrict__ c1_b,
    const float* __restrict__ c1_g, const float* __restrict__ c1_be,
    const _Float16* __restrict__ c2P, const float* __restrict__ c2_b,
    const float* __restrict__ c3_w, const float* __restrict__ c3_b,
    float* __restrict__ out, int Btot) {
  __shared__ __align__(16) char smem[LDS_SZ];
  float*    XAF0 = (float*)(smem + OFF_XAF0);
  float*    XAF1 = (float*)(smem + OFF_XAF1);
  _Float16* XB0  = (_Float16*)(smem + OFF_XB0);
  _Float16* XB1  = (_Float16*)(smem + OFF_XB1);
  float*    S32  = (float*)(smem + OFF_S32);
  _Float16* CB   = (_Float16*)(smem + OFF_CB);
  _Float16* H1   = (_Float16*)(smem + OFF_H1);
  _Float16* H2   = (_Float16*)(smem + OFF_H2);
  _Float16* HM   = (_Float16*)(smem + OFF_HM);
  _Float16* HC   = (_Float16*)(smem + OFF_HC);
  float*    RS   = (float*)(smem + OFF_RS);

  const int tid = threadIdx.x;
  const int lane = tid & 31, wv = tid >> 5;
  const int rowsub = wv >> 1, nhalf = wv & 1;
  const int m = lane & 15;                 // A row / C column within tile
  const int kh8 = (lane >> 4) * 16;        // A K-half select
  const int rh8 = (lane >> 4) * 8;         // C row offset
  const size_t tile0 = (size_t)blockIdx.x * ROWS;

  // stage chunk ch of (x-rows, packed weights) into given buffers (async)
  auto stage = [&](int ch, int which, const _Float16* WP, float* XAF,
                   _Float16* XBc) {
    const int k0 = ch * KC;
    // A: 64 rows x 160 f32 = 2560 x 16B segments
#pragma unroll
    for (int i = 0; i < 10; ++i) {
      int e = i * 256 + tid;
      int r = e / 40, seg = e - r * 40;
      int kk = k0 + seg * 4;
      if (kk < IN_DIM) {
        const float* gp = x + ((tile0 + r) * 2 + which) * (size_t)IN_DIM + kk;
        async_b128(gp, XAF + r * XAF_LD + seg * 4);
      } else {                               // zero-pad K in [2056,2080)
        float4 z = make_float4(0.f, 0.f, 0.f, 0.f);
        *(float4*)(XAF + r * XAF_LD + seg * 4) = z;
      }
    }
    // B: contiguous 80KB span of packed tiles
    const char* gs = (const char*)(WP + (size_t)(k0 >> 5) * 16 * 512);
    char* ls = (char*)XBc;
#pragma unroll
    for (int i = 0; i < 20; ++i) {
      int o = (i * 256 + tid) * 16;
      async_b128(gs + o, ls + o);
    }
  };

  auto compute_chunk = [&](const float* XAF, const _Float16* XBc, v8f* acc) {
    const float* Arow = XAF + (rowsub * 16) * XAF_LD;
#pragma unroll
    for (int s = 0; s < 5; ++s) {
      v16h a = load_a_f32(Arow, XAF_LD, m, s * 32, kh8);
      v16h bf[8];
#pragma unroll
      for (int j = 0; j < 8; ++j)
        bf[j] = *(const v16h*)(XBc + (size_t)(s * 16 + nhalf * 8 + j) * 512 +
                               lane * 16);
#pragma unroll
      for (int j = 0; j < 8; ++j) acc[j] = WMMA(a, bf[j], acc[j]);
    }
  };

  // ======================= G1/G2: embeds + LN + relu =======================
  for (int which = 0; which < 2; ++which) {
    const _Float16* WP = which ? ceP : meP;
    const float* bias = which ? ce_b : me_b;
    const float* g    = which ? ce_g : me_g;
    const float* be   = which ? ce_be : me_be;
    _Float16* HOUT    = which ? HC : HM;

    v8f acc[8];
#pragma unroll
    for (int j = 0; j < 8; ++j) {
      float bv = bias[nhalf * 128 + j * 16 + m];
#pragma unroll
      for (int r = 0; r < 8; ++r) acc[j][r] = bv;
    }

    stage(0, which, WP, XAF0, XB0);          // prologue
    for (int ch = 0; ch < NCHUNK; ++ch) {
      const int cur = ch & 1;
      wait_async0();                         // this wave's chunk-ch DMA done
      __syncthreads();                       // all waves: buf cur valid,
                                             // buf 1-cur no longer being read
      if (ch + 1 < NCHUNK)                   // overlap next DMA with compute
        stage(ch + 1, which, WP, cur ? XAF0 : XAF1, cur ? XB0 : XB1);
      compute_chunk(cur ? XAF1 : XAF0, cur ? XB1 : XB0, acc);
    }
    __syncthreads();                         // S32 aliases XAF: all reads done

    // stage fp32 result for LN
#pragma unroll
    for (int j = 0; j < 8; ++j) {
      int col = nhalf * 128 + j * 16 + m;
#pragma unroll
      for (int r = 0; r < 8; ++r)
        S32[(rowsub * 16 + r + rh8) * S32_LD + col] = acc[j][r];
    }
    __syncthreads();
    if (tid < 64) {
      float s = 0.0f, s2 = 0.0f;
      for (int c = 0; c < 256; ++c) {
        float v = S32[tid * S32_LD + c];
        s += v; s2 += v * v;
      }
      float mu = s * (1.0f / 256.0f);
      float var = s2 * (1.0f / 256.0f) - mu * mu;
      RS[tid * 2] = mu;
      RS[tid * 2 + 1] = rsqrtf(var + 1e-5f);
    }
    __syncthreads();
    {
      float gc = g[tid], bc = be[tid];
      for (int r = 0; r < 64; ++r) {
        float v = (S32[r * S32_LD + tid] - RS[r * 2]) * RS[r * 2 + 1] * gc + bc;
        HOUT[r * H_LD + tid] = (_Float16)fmaxf(v, 0.0f);
      }
    }
    __syncthreads();
  }

  // ========== G3/G4: fused attention linears -> combined (f16) ============
  for (int pass = 0; pass < 2; ++pass) {
    const _Float16* A  = pass ? HM : HC;   // cross-att: h_ctx, self: h_mut
    const _Float16* WP = pass ? saP : caP;
    const float*   bvo = pass ? sab : cab;
    v8f acc[8];
#pragma unroll
    for (int j = 0; j < 8; ++j) {
      float bv = bvo[nhalf * 128 + j * 16 + m];
#pragma unroll
      for (int r = 0; r < 8; ++r) acc[j][r] = bv;
    }
    const _Float16* Arow = A + (rowsub * 16) * H_LD;
#pragma unroll
    for (int s = 0; s < 8; ++s) {
      v16h a = load_a(Arow, H_LD, m, s * 32, kh8);
      v16h bf[8];
#pragma unroll
      for (int j = 0; j < 8; ++j)
        bf[j] = *(const v16h*)(WP + (size_t)(s * 16 + nhalf * 8 + j) * 512 +
                               lane * 16);
#pragma unroll
      for (int j = 0; j < 8; ++j) acc[j] = WMMA(a, bf[j], acc[j]);
    }
#pragma unroll
    for (int j = 0; j < 8; ++j) {
      int col = pass * 256 + nhalf * 128 + j * 16 + m;
#pragma unroll
      for (int r = 0; r < 8; ++r)
        CB[(size_t)(rowsub * 16 + r + rh8) * CB_LD + col] = (_Float16)acc[j][r];
    }
  }
  __syncthreads();

  // ================= G5: combined(512) @ c1 -> LN -> relu =================
  {
    v8f acc[8];
#pragma unroll
    for (int j = 0; j < 8; ++j) {
      float bv = c1_b[nhalf * 128 + j * 16 + m];
#pragma unroll
      for (int r = 0; r < 8; ++r) acc[j][r] = bv;
    }
    const _Float16* Arow = CB + (size_t)(rowsub * 16) * CB_LD;
#pragma unroll
    for (int s = 0; s < 16; ++s) {
      v16h a = load_a(Arow, CB_LD, m, s * 32, kh8);
      v16h bf[8];
#pragma unroll
      for (int j = 0; j < 8; ++j)
        bf[j] = *(const v16h*)(c1P + (size_t)(s * 16 + nhalf * 8 + j) * 512 +
                               lane * 16);
#pragma unroll
      for (int j = 0; j < 8; ++j) acc[j] = WMMA(a, bf[j], acc[j]);
    }
    __syncthreads();                         // S32 aliases CB: all reads done
#pragma unroll
    for (int j = 0; j < 8; ++j) {
      int col = nhalf * 128 + j * 16 + m;
#pragma unroll
      for (int r = 0; r < 8; ++r)
        S32[(rowsub * 16 + r + rh8) * S32_LD + col] = acc[j][r];
    }
    __syncthreads();
    if (tid < 64) {
      float s = 0.0f, s2 = 0.0f;
      for (int c = 0; c < 256; ++c) {
        float v = S32[tid * S32_LD + c];
        s += v; s2 += v * v;
      }
      float mu = s * (1.0f / 256.0f);
      float var = s2 * (1.0f / 256.0f) - mu * mu;
      RS[tid * 2] = mu;
      RS[tid * 2 + 1] = rsqrtf(var + 1e-5f);
    }
    __syncthreads();
    {
      float gc = c1_g[tid], bc = c1_be[tid];
      for (int r = 0; r < 64; ++r) {
        float v = (S32[r * S32_LD + tid] - RS[r * 2]) * RS[r * 2 + 1] * gc + bc;
        H1[r * H_LD + tid] = (_Float16)fmaxf(v, 0.0f);
      }
    }
    __syncthreads();
  }

  // ==================== G6: h1(256) @ c2 -> relu (64) =====================
  {
    v8f acc[2];
    const int ntb = nhalf * 2;
#pragma unroll
    for (int j = 0; j < 2; ++j) {
      float bv = c2_b[(ntb + j) * 16 + m];
#pragma unroll
      for (int r = 0; r < 8; ++r) acc[j][r] = bv;
    }
    const _Float16* Arow = H1 + (rowsub * 16) * H_LD;
#pragma unroll
    for (int s = 0; s < 8; ++s) {
      v16h a = load_a(Arow, H_LD, m, s * 32, kh8);
      v16h bf[2];
#pragma unroll
      for (int j = 0; j < 2; ++j)
        bf[j] = *(const v16h*)(c2P + (size_t)(s * 4 + ntb + j) * 512 + lane * 16);
#pragma unroll
      for (int j = 0; j < 2; ++j) acc[j] = WMMA(a, bf[j], acc[j]);
    }
#pragma unroll
    for (int j = 0; j < 2; ++j) {
      int col = (ntb + j) * 16 + m;
#pragma unroll
      for (int r = 0; r < 8; ++r)
        H2[(rowsub * 16 + r + rh8) * H2_LD + col] =
            (_Float16)fmaxf(acc[j][r], 0.0f);
    }
    __syncthreads();
  }

  // ==================== G7: h2(64) @ c3 -> out (1) ========================
  if (tid < 64) {
    size_t row = tile0 + tid;
    if (row < (size_t)Btot) {
      float acc = c3_b[0];
      for (int j = 0; j < 64; ++j)
        acc += (float)H2[tid * H2_LD + j] * c3_w[j];
      out[row] = acc;
    }
  }
}

// ---------------------------------------------------------------------------
extern "C" void kernel_launch(void* const* d_in, const int* in_sizes, int n_in,
                              void* d_out, int out_size, void* d_ws, size_t ws_size,
                              hipStream_t stream) {
  const float* x       = (const float*)d_in[0];
  const float* me_w    = (const float*)d_in[1];
  const float* me_b    = (const float*)d_in[2];
  const float* me_g    = (const float*)d_in[3];
  const float* me_be   = (const float*)d_in[4];
  const float* ce_w    = (const float*)d_in[5];
  const float* ce_b    = (const float*)d_in[6];
  const float* ce_g    = (const float*)d_in[7];
  const float* ce_be   = (const float*)d_in[8];
  const float* ca_in_w = (const float*)d_in[9];
  const float* ca_in_b = (const float*)d_in[10];
  const float* ca_out_w= (const float*)d_in[11];
  const float* ca_out_b= (const float*)d_in[12];
  const float* sa_in_w = (const float*)d_in[13];
  const float* sa_in_b = (const float*)d_in[14];
  const float* sa_out_w= (const float*)d_in[15];
  const float* sa_out_b= (const float*)d_in[16];
  const float* c1_w    = (const float*)d_in[17];
  const float* c1_b    = (const float*)d_in[18];
  const float* c1_g    = (const float*)d_in[19];
  const float* c1_be   = (const float*)d_in[20];
  const float* c2_w    = (const float*)d_in[21];
  const float* c2_b    = (const float*)d_in[22];
  const float* c3_w    = (const float*)d_in[23];
  const float* c3_b    = (const float*)d_in[24];

  const int Btot = in_sizes[0] / (2 * IN_DIM);

  char* ws = (char*)d_ws;
  size_t off = 0;
  _Float16* meP = (_Float16*)(ws + off); off += (size_t)KPAD * 256 * 2;
  _Float16* ceP = (_Float16*)(ws + off); off += (size_t)KPAD * 256 * 2;
  _Float16* caP = (_Float16*)(ws + off); off += (size_t)256 * 256 * 2;
  _Float16* saP = (_Float16*)(ws + off); off += (size_t)256 * 256 * 2;
  _Float16* c1P = (_Float16*)(ws + off); off += (size_t)512 * 256 * 2;
  _Float16* c2P = (_Float16*)(ws + off); off += (size_t)256 * 64 * 2;
  float*    cab = (float*)(ws + off);    off += 256 * 4;
  float*    sab = (float*)(ws + off);    off += 256 * 4;

  // --- weight prep (every call; deterministic) ---
  pack_w16<<<(KPAD * 256 + 255) / 256, 256, 0, stream>>>(me_w, meP, IN_DIM, KPAD, 256);
  pack_w16<<<(KPAD * 256 + 255) / 256, 256, 0, stream>>>(ce_w, ceP, IN_DIM, KPAD, 256);
  pack_w16<<<(512 * 256 + 255) / 256, 256, 0, stream>>>(c1_w, c1P, 512, 512, 256);
  pack_w16<<<(256 * 64 + 255) / 256, 256, 0, stream>>>(c2_w, c2P, 256, 256, 64);
  fuse_vo_pack<<<256, 256, 0, stream>>>(ca_in_w, ca_out_w, caP);
  fuse_vo_pack<<<256, 256, 0, stream>>>(sa_in_w, sa_out_w, saP);
  fuse_vo_bias<<<1, 256, 0, stream>>>(ca_in_b, ca_out_w, ca_out_b, cab);
  fuse_vo_bias<<<1, 256, 0, stream>>>(sa_in_b, sa_out_w, sa_out_b, sab);

  // --- fused forward ---
  fused_net<<<(Btot + ROWS - 1) / ROWS, 256, 0, stream>>>(
      x, meP, me_b, me_g, me_be, ceP, ce_b, ce_g, ce_be,
      caP, cab, saP, sab, c1P, c1_b, c1_g, c1_be,
      c2P, c2_b, c3_w, c3_b, (float*)d_out, Btot);
}
